// TGAT_36026185678965
// MI455X (gfx1250) — matrix operand order repaired and checked by reading, not compiled, not run on hardware
//
#include <hip/hip_runtime.h>
#include <math.h>

typedef __attribute__((ext_vector_type(16))) _Float16 v16h;
typedef __attribute__((ext_vector_type(8)))  _Float16 v8h;
typedef __attribute__((ext_vector_type(4)))  _Float16 v4h;
typedef __attribute__((ext_vector_type(8)))  float    v8f;

#define NN   40000      // nodes
#define NE   1000000    // edges
#define FIN_ 128
#define HID_ 128
#define TD   16
#define HEADS_ 4
#define CH   32         // per-head channels
#define NG   64         // graphs
#define E2   (NE + NN)  // edges incl. self loops
#define NEG_SLOPE 0.2f
#define LN_EPS 1e-5f
#define CDIV(a,b) (((a)+(b)-1)/(b))

// ---------- order-preserving float<->uint map (for atomic float max) -------
static __device__ __forceinline__ unsigned f2ord(float f) {
  unsigned u = __float_as_uint(f);
  return (u & 0x80000000u) ? ~u : (u | 0x80000000u);
}
static __device__ __forceinline__ float ord2f(unsigned u) {
  return (u & 0x80000000u) ? __uint_as_float(u & 0x7fffffffu)
                           : __uint_as_float(~u);
}

// ---------------------------- zero fill ------------------------------------
__global__ void k_zero(unsigned* __restrict__ p, long long n) {
  long long i = (long long)blockIdx.x * blockDim.x + threadIdx.x;
  long long st = (long long)gridDim.x * blockDim.x;
  for (; i < n; i += st) p[i] = 0u;
}

// ------------- WMMA GEMM: D[M,128] = A[M,K] * W[128,K]^T + bias ------------
// K and LDD compile-time -> fully unrolled branch-free WMMA chain and
// immediate-offset stores (no per-row 64-bit address arithmetic).
// block = 256 = 8 waves; wave = 16-col N-tile; block strides over M-tiles.
// Weight fragments hoisted out of the M-loop; A tile staged via LDS (f16).
// Fragment layouts per CDNA5 ISA 7.12.2 (A 8/8 K-split, B 8/8 K-split).
template <int K, int LDD>
__global__ void k_gemm_wmma(const float* __restrict__ A,
                            const float* __restrict__ W,
                            const float* __restrict__ bias,
                            float* __restrict__ D, int mtiles) {
  constexpr int KB = (K + 31) / 32;                  // 32-wide k-steps
  __shared__ alignas(16) _Float16 aT[16 * K];        // staged A tile (f16)
  const int tid   = threadIdx.x;
  const int lane  = tid & 31;
  const int ntile = tid >> 5;                        // 0..7
  const int half  = lane >> 4;                       // 0/1
  const int l15   = lane & 15;
  const int col   = ntile * 16 + l15;                // output column / W row

  // ---- hoisted weight fragments (reused by every M-tile) ----
  v16h bf[KB];
  {
    const float* __restrict__ Wr = W + (size_t)col * K;
    #pragma unroll
    for (int kb = 0; kb < KB; ++kb) {
      const int kq = kb * 32 + half * 8;             // B lanes: 8/8 K runs
      if (kq + 8 <= K) {
        #pragma unroll
        for (int q = 0; q < 2; ++q) {
          float4 f = *(const float4*)(Wr + kq + q * 4);
          bf[kb][q * 4 + 0] = (_Float16)f.x;
          bf[kb][q * 4 + 1] = (_Float16)f.y;
          bf[kb][q * 4 + 2] = (_Float16)f.z;
          bf[kb][q * 4 + 3] = (_Float16)f.w;
        }
      } else {
        #pragma unroll
        for (int j = 0; j < 8; ++j) bf[kb][j] = (_Float16)0.f;
      }
      if (kq + 24 <= K) {
        #pragma unroll
        for (int q = 0; q < 2; ++q) {
          float4 f = *(const float4*)(Wr + kq + 16 + q * 4);
          bf[kb][8 + q * 4 + 0] = (_Float16)f.x;
          bf[kb][8 + q * 4 + 1] = (_Float16)f.y;
          bf[kb][8 + q * 4 + 2] = (_Float16)f.z;
          bf[kb][8 + q * 4 + 3] = (_Float16)f.w;
        }
      } else {
        #pragma unroll
        for (int j = 0; j < 8; ++j) bf[kb][8 + j] = (_Float16)0.f;
      }
    }
  }
  const float bv = bias ? bias[col] : 0.f;

  for (int m = blockIdx.x; m < mtiles; m += (int)gridDim.x) {
    const float* __restrict__ Ab = A + (size_t)m * 16 * K;  // contiguous tile
    __builtin_prefetch(Ab + (size_t)gridDim.x * 16 * K, 0, 0); // next tile
    __syncthreads();                                 // aT reads done
    #pragma unroll 1
    for (int i = tid; i < (16 * K) / 4; i += 256) {  // coalesced f32->f16
      float4 f = ((const float4*)Ab)[i];
      v4h h4 = { (_Float16)f.x, (_Float16)f.y, (_Float16)f.z, (_Float16)f.w };
      ((v4h*)aT)[i] = h4;
    }
    __syncthreads();

    v8f acc = {};
    const int abase = l15 * K;                       // this lane's A row
    #pragma unroll
    for (int kb = 0; kb < KB; ++kb) {
      v16h a;
      if (kb * 32 + 16 <= K) {                       // compile-time guard
        v8h c = *(const v8h*)(aT + abase + kb * 32 + half * 8);
        #pragma unroll
        for (int j = 0; j < 8; ++j) a[j] = c[j];
      } else {
        #pragma unroll
        for (int j = 0; j < 8; ++j) a[j] = (_Float16)0.f;
      }
      if (kb * 32 + 32 <= K) {                       // compile-time guard
        v8h c = *(const v8h*)(aT + abase + kb * 32 + 16 + half * 8);
        #pragma unroll
        for (int j = 0; j < 8; ++j) a[8 + j] = c[j];
      } else {
        #pragma unroll
        for (int j = 0; j < 8; ++j) a[8 + j] = (_Float16)0.f;
      }
      acc = __builtin_amdgcn_wmma_f32_16x16x32_f16(false, a, false, bf[kb],
                                                   (short)0, acc, false, false);
    }
    // C/D layout: VGPR r -> row r (lanes 0-15) / row 8+r (lanes 16-31).
    // Single base pointer; r*LDD folds into the store's immediate offset.
    float* __restrict__ Dp = D + (size_t)(m * 16 + half * 8) * LDD + col;
    #pragma unroll
    for (int r = 0; r < 8; ++r)
      Dp[r * LDD] = acc[r] + bv;
  }
}

// ------------------ time encoding scatter-add onto dst ---------------------
__global__ void k_time(const float* __restrict__ ts, const int* __restrict__ dst,
                       const float* __restrict__ tw, const float* __restrict__ tb,
                       float* __restrict__ tnode) {
  int e = blockIdx.x * blockDim.x + threadIdx.x;
  if (e >= NE) return;
  float t = ts[e];
  float* out = tnode + (size_t)dst[e] * TD;
  #pragma unroll
  for (int j = 0; j < TD; ++j)
    atomicAdd(&out[j], __cosf(t * tw[j] + tb[j]));
}

__global__ void k_concat_t(const float* __restrict__ tnode, float* __restrict__ hcat) {
  int i = blockIdx.x * blockDim.x + threadIdx.x;   // N*TD
  if (i >= NN * TD) return;
  int n = i / TD, j = i % TD;
  hcat[(size_t)n * (HID_ + TD) + HID_ + j] = tnode[i];
}

// ------------------ per-node attention logits ------------------------------
__global__ void k_att(const float* __restrict__ xh, const float* __restrict__ atts,
                      const float* __restrict__ attd,
                      float* __restrict__ a_src, float* __restrict__ a_dst) {
  int i = blockIdx.x * blockDim.x + threadIdx.x;   // N*HEADS
  if (i >= NN * HEADS_) return;
  int n = i >> 2, h = i & 3;
  const float* xp = xh + (size_t)n * HID_ + h * CH;
  const float* sp = atts + h * CH;
  const float* dp = attd + h * CH;
  float ss = 0.f, sd = 0.f;
  #pragma unroll
  for (int c = 0; c < CH; ++c) { float v = xp[c]; ss += v * sp[c]; sd += v * dp[c]; }
  a_src[i] = ss; a_dst[i] = sd;
}

static __device__ __forceinline__ void edge_sd(const int* src, const int* dst,
                                               int e, int& s, int& d) {
  if (e < NE) { s = src[e]; d = dst[e]; } else { s = e - NE; d = s; }  // self loop
}

__global__ void k_edge_max(const int* __restrict__ src, const int* __restrict__ dst,
                           const float* __restrict__ a_src, const float* __restrict__ a_dst,
                           unsigned* __restrict__ mkey) {
  int i = blockIdx.x * blockDim.x + threadIdx.x;   // E2*HEADS
  if (i >= E2 * HEADS_) return;
  int e = i >> 2, h = i & 3, s, d;
  edge_sd(src, dst, e, s, d);
  float el = a_src[s * HEADS_ + h] + a_dst[d * HEADS_ + h];
  el = el > 0.f ? el : NEG_SLOPE * el;
  atomicMax(&mkey[d * HEADS_ + h], f2ord(el));
}

__global__ void k_edge_sum(const int* __restrict__ src, const int* __restrict__ dst,
                           const float* __restrict__ a_src, const float* __restrict__ a_dst,
                           const unsigned* __restrict__ mkey, float* __restrict__ denom) {
  int i = blockIdx.x * blockDim.x + threadIdx.x;
  if (i >= E2 * HEADS_) return;
  int e = i >> 2, h = i & 3, s, d;
  edge_sd(src, dst, e, s, d);
  int dh = d * HEADS_ + h;
  float el = a_src[s * HEADS_ + h] + a_dst[dh];
  el = el > 0.f ? el : NEG_SLOPE * el;
  atomicAdd(&denom[dh], __expf(el - ord2f(mkey[dh])));
}

__global__ void k_edge_msg(const int* __restrict__ src, const int* __restrict__ dst,
                           const float* __restrict__ a_src, const float* __restrict__ a_dst,
                           const unsigned* __restrict__ mkey, const float* __restrict__ denom,
                           const float* __restrict__ xh, float* __restrict__ agg) {
  int i = blockIdx.x * blockDim.x + threadIdx.x;
  if (i >= E2 * HEADS_) return;
  int e = i >> 2, h = i & 3, s, d;
  edge_sd(src, dst, e, s, d);
  int dh = d * HEADS_ + h;
  float el = a_src[s * HEADS_ + h] + a_dst[dh];
  el = el > 0.f ? el : NEG_SLOPE * el;
  float alpha = __expf(el - ord2f(mkey[dh])) / denom[dh];
  const float4* xp = (const float4*)(xh + (size_t)s * HID_ + h * CH);
  float* op = agg + (size_t)d * HID_ + h * CH;
  #pragma unroll
  for (int q = 0; q < CH / 4; ++q) {
    float4 v = xp[q];
    atomicAdd(&op[q * 4 + 0], v.x * alpha);
    atomicAdd(&op[q * 4 + 1], v.y * alpha);
    atomicAdd(&op[q * 4 + 2], v.z * alpha);
    atomicAdd(&op[q * 4 + 3], v.w * alpha);
  }
}

// ------------- fused bias + LayerNorm + ReLU (one block per node) ----------
__global__ void k_lnrelu(const float* __restrict__ agg, const float* __restrict__ gb,
                         const float* __restrict__ lw, const float* __restrict__ lb,
                         float* __restrict__ out) {
  __shared__ float s1[HID_], s2[HID_];
  int n = blockIdx.x, c = threadIdx.x;  // blockDim = 128
  float v = agg[(size_t)n * HID_ + c] + gb[c];
  s1[c] = v; s2[c] = v * v;
  __syncthreads();
  for (int st = HID_ / 2; st > 0; st >>= 1) {
    if (c < st) { s1[c] += s1[c + st]; s2[c] += s2[c + st]; }
    __syncthreads();
  }
  float mu  = s1[0] * (1.f / HID_);
  float var = s2[0] * (1.f / HID_) - mu * mu;
  float r   = rsqrtf(var + LN_EPS);
  float o   = (v - mu) * r * lw[c] + lb[c];
  out[(size_t)n * HID_ + c] = o > 0.f ? o : 0.f;
}

// ----------------------------- pooling -------------------------------------
__global__ void k_pool(const float* __restrict__ h, const int* __restrict__ batch,
                       float* __restrict__ psum, unsigned* __restrict__ pmax,
                       float* __restrict__ pcnt) {
  int i = blockIdx.x * blockDim.x + threadIdx.x;   // N*HID
  if (i >= NN * HID_) return;
  int n = i >> 7, c = i & 127;
  int g = batch[n];
  float v = h[i];
  atomicAdd(&psum[g * HID_ + c], v);
  atomicMax(&pmax[g * HID_ + c], f2ord(v));
  if (c == 0) atomicAdd(&pcnt[g], 1.f);
}

__global__ void k_hg(const float* __restrict__ psum, const unsigned* __restrict__ pmax,
                     const float* __restrict__ pcnt, float* __restrict__ hg) {
  int i = blockIdx.x * blockDim.x + threadIdx.x;   // G*2*HID
  if (i >= NG * 2 * HID_) return;
  int g = i / (2 * HID_), c = i % (2 * HID_);
  hg[i] = (c < HID_) ? psum[g * HID_ + c] / pcnt[g]
                     : ord2f(pmax[g * HID_ + c - HID_]);
}

// ----------------------------- classifier ----------------------------------
__global__ void k_cls1(const float* __restrict__ hg, const float* __restrict__ w,
                       const float* __restrict__ b, float* __restrict__ out) {
  int i = blockIdx.x * blockDim.x + threadIdx.x;   // G*HID
  if (i >= NG * HID_) return;
  int g = i >> 7, j = i & 127;
  const float* hp = hg + (size_t)g * 2 * HID_;
  const float* wp = w + (size_t)j * 2 * HID_;
  float s = b[j];
  for (int k = 0; k < 2 * HID_; ++k) s += hp[k] * wp[k];
  out[i] = s > 0.f ? s : 0.f;
}

__global__ void k_cls2(const float* __restrict__ hc, const float* __restrict__ w,
                       const float* __restrict__ b, float* __restrict__ out) {
  int i = blockIdx.x * blockDim.x + threadIdx.x;   // G*2
  if (i >= NG * 2) return;
  int g = i >> 1, j = i & 1;
  const float* hp = hc + (size_t)g * HID_;
  const float* wp = w + (size_t)j * HID_;
  float s = b[j];
  for (int k = 0; k < HID_; ++k) s += hp[k] * wp[k];
  out[i] = s;
}

// ---------------------------------------------------------------------------
extern "C" void kernel_launch(void* const* d_in, const int* in_sizes, int n_in,
                              void* d_out, int out_size, void* d_ws, size_t ws_size,
                              hipStream_t stream) {
  (void)in_sizes; (void)n_in; (void)out_size; (void)ws_size;
  const float* x      = (const float*)d_in[0];
  const int*   eidx   = (const int*)d_in[1];
  const float* ts     = (const float*)d_in[2];
  const int*   batch  = (const int*)d_in[3];
  const float* time_w = (const float*)d_in[4];
  const float* time_b = (const float*)d_in[5];
  const float* proj_w = (const float*)d_in[6];
  const float* proj_b = (const float*)d_in[7];
  const float* gat0_w = (const float*)d_in[8];
  const float* atts0  = (const float*)d_in[9];
  const float* attd0  = (const float*)d_in[10];
  const float* gat0_b = (const float*)d_in[11];
  const float* ln0_w  = (const float*)d_in[12];
  const float* ln0_b  = (const float*)d_in[13];
  const float* gat1_w = (const float*)d_in[14];
  const float* atts1  = (const float*)d_in[15];
  const float* attd1  = (const float*)d_in[16];
  const float* gat1_b = (const float*)d_in[17];
  const float* ln1_w  = (const float*)d_in[18];
  const float* ln1_b  = (const float*)d_in[19];
  const float* clw1   = (const float*)d_in[20];
  const float* clb1   = (const float*)d_in[21];
  const float* clw2   = (const float*)d_in[22];
  const float* clb2   = (const float*)d_in[23];
  const int* srcA = eidx;
  const int* dstA = eidx + NE;
  float* out = (float*)d_out;

  // ---- workspace layout (floats) ----
  float* ws    = (float*)d_ws;
  float* hcat  = ws;                       // [N, 144]   (layer-0 GEMM input)
  float* hmid  = hcat;                     // [N, 128]   alias: hcat dead after gemm
  float* xh    = hcat + (size_t)NN * (HID_ + TD);   // [N, 128]
  float* agg   = xh   + (size_t)NN * HID_;          // [N, 128]
  float* tnode = agg  + (size_t)NN * HID_;          // [N, 16]
  float* a_src = tnode + (size_t)NN * TD;           // [N, 4]
  float* a_dst = a_src + (size_t)NN * HEADS_;       // [N, 4]
  unsigned* mkey = (unsigned*)(a_dst + (size_t)NN * HEADS_);  // [N, 4]
  float* denom = (float*)(mkey + (size_t)NN * HEADS_);        // [N, 4]
  float* psum  = denom + (size_t)NN * HEADS_;       // [G, 128]
  unsigned* pmax = (unsigned*)(psum + NG * HID_);   // [G, 128]
  float* pcnt  = (float*)(pmax + NG * HID_);        // [G]
  float* hg    = pcnt + NG;                         // [G, 256]
  float* hcls  = hg + NG * 2 * HID_;                // [G, 128]

  const int B  = 256;
  const int ZG = 2048;
  const int GG = 640;                // persistent GEMM blocks
  const int mt = NN / 16;            // 2500 M-tiles
  const int eg = CDIV(E2 * HEADS_, B);

  // ---- time encoding -> tnode ----
  k_zero<<<ZG, B, 0, stream>>>((unsigned*)tnode, (long long)NN * TD);
  k_time<<<CDIV(NE, B), B, 0, stream>>>(ts, dstA, time_w, time_b, tnode);

  // ---- node projection (WMMA) + concat time features ----
  k_gemm_wmma<FIN_, HID_ + TD><<<GG, B, 0, stream>>>(x, proj_w, proj_b, hcat, mt);
  k_concat_t<<<CDIV(NN * TD, B), B, 0, stream>>>(tnode, hcat);

  // ================= GAT layer 0 =================
  k_gemm_wmma<HID_ + TD, HID_><<<GG, B, 0, stream>>>(hcat, gat0_w, nullptr, xh, mt);
  k_att<<<CDIV(NN * HEADS_, B), B, 0, stream>>>(xh, atts0, attd0, a_src, a_dst);
  k_zero<<<ZG, B, 0, stream>>>(mkey, (long long)NN * HEADS_);
  k_zero<<<ZG, B, 0, stream>>>((unsigned*)denom, (long long)NN * HEADS_);
  k_zero<<<ZG, B, 0, stream>>>((unsigned*)agg, (long long)NN * HID_);
  k_edge_max<<<eg, B, 0, stream>>>(srcA, dstA, a_src, a_dst, mkey);
  k_edge_sum<<<eg, B, 0, stream>>>(srcA, dstA, a_src, a_dst, mkey, denom);
  k_edge_msg<<<eg, B, 0, stream>>>(srcA, dstA, a_src, a_dst, mkey, denom, xh, agg);
  k_lnrelu<<<NN, HID_, 0, stream>>>(agg, gat0_b, ln0_w, ln0_b, hmid);

  // ================= GAT layer 1 =================
  k_gemm_wmma<HID_, HID_><<<GG, B, 0, stream>>>(hmid, gat1_w, nullptr, xh, mt);
  k_att<<<CDIV(NN * HEADS_, B), B, 0, stream>>>(xh, atts1, attd1, a_src, a_dst);
  k_zero<<<ZG, B, 0, stream>>>(mkey, (long long)NN * HEADS_);
  k_zero<<<ZG, B, 0, stream>>>((unsigned*)denom, (long long)NN * HEADS_);
  k_zero<<<ZG, B, 0, stream>>>((unsigned*)agg, (long long)NN * HID_);
  k_edge_max<<<eg, B, 0, stream>>>(srcA, dstA, a_src, a_dst, mkey);
  k_edge_sum<<<eg, B, 0, stream>>>(srcA, dstA, a_src, a_dst, mkey, denom);
  k_edge_msg<<<eg, B, 0, stream>>>(srcA, dstA, a_src, a_dst, mkey, denom, xh, agg);
  k_lnrelu<<<NN, HID_, 0, stream>>>(agg, gat1_b, ln1_w, ln1_b, hmid);

  // ================= pooling + classifier =================
  k_zero<<<ZG, B, 0, stream>>>((unsigned*)psum, NG * HID_);
  k_zero<<<ZG, B, 0, stream>>>(pmax, NG * HID_);
  k_zero<<<ZG, B, 0, stream>>>((unsigned*)pcnt, NG);
  k_pool<<<CDIV(NN * HID_, B), B, 0, stream>>>(hmid, batch, psum, pmax, pcnt);
  k_hg<<<CDIV(NG * 2 * HID_, B), B, 0, stream>>>(psum, pmax, pcnt, hg);
  k_cls1<<<CDIV(NG * HID_, B), B, 0, stream>>>(hg, clw1, clb1, hcls);
  k_cls2<<<CDIV(NG * 2, B), B, 0, stream>>>(hcls, clw2, clb2, out);
}
